// GMM_88433376625119
// MI455X (gfx1250) — matrix operand order repaired and checked by reading, not compiled, not run on hardware
//
#include <hip/hip_runtime.h>
#include <hip/hip_bf16.h>
#include <cstdint>
#include <cstddef>

// ---------------------------------------------------------------------------
// GMM forward for MI455X (gfx1250): f16 NHWC activations, WMMA implicit-GEMM
// convs (v_wmma_f32_16x16x32_f16), BN folded into packed weights, weights
// staged into LDS by the Tensor Data Mover (double-buffered over taps).
// ---------------------------------------------------------------------------

#ifndef GMM_USE_TDM
#define GMM_USE_TDM 1
#endif

typedef _Float16 h16 __attribute__((ext_vector_type(16)));  // WMMA A/B operand (8 VGPRs)
typedef _Float16 h8  __attribute__((ext_vector_type(8)));   // 16-byte half vector
typedef float    f8  __attribute__((ext_vector_type(8)));   // WMMA C/D operand (8 VGPRs)

typedef unsigned int tdm_u32x4 __attribute__((ext_vector_type(4)));
typedef int          tdm_i32x4 __attribute__((ext_vector_type(4)));
typedef int          tdm_i32x8 __attribute__((ext_vector_type(8)));

#define BN_EPS 1e-5f

__device__ __forceinline__ h16 h16_zero() {
  h16 r;
#pragma unroll
  for (int i = 0; i < 16; ++i) r[i] = (_Float16)0.0f;
  return r;
}

#if GMM_USE_TDM
// ---------------------------------------------------------------------------
// Issue a 1-D TDM copy: nelem 2-byte elements from global 'gsrc' to LDS byte
// offset 'lds_off'. Descriptor packed per CDNA5 ISA ch.8 (D# groups 0/1;
// remaining groups zero = dims 2+ unused). Tracked by TENSORcnt.
// This toolchain exposes the 6-arg builtin:
//   (u32x4 g0, i32x8 g1, i32x4, i32x4, i32x8, i32 cpol)
// ---------------------------------------------------------------------------
__device__ __forceinline__ void tdm_load_1d(const void* gsrc, unsigned lds_off,
                                            unsigned nelem) {
  const unsigned long long ga = (unsigned long long)(uintptr_t)gsrc;
  tdm_u32x4 g0;
  g0[0] = 1u;                                                   // count=1 (valid user D#)
  g0[1] = lds_off;                                              // lds_addr (bytes)
  g0[2] = (unsigned)ga;                                         // global_addr[31:0]
  g0[3] = (unsigned)((ga >> 32) & 0x1FFFFFFull) | (2u << 30);   // addr[56:32] | type=2

  // group1 (256 bits) packed via four 64-bit lanes
  const unsigned long long dim0 = (unsigned long long)nelem;
  unsigned long long q0 = (1ull << 16)                 // data_size = 1 -> 2 bytes
                        | ((dim0 & 0xFFFFull) << 48);  // tensor_dim0[15:0]  (bits 63:48)
  unsigned long long q1 = ((dim0 >> 16) & 0xFFFFull)   // tensor_dim0[31:16] (bits 79:64)
                        | (1ull << 16)                 // tensor_dim1 = 1    (bits 111:80)
                        | ((dim0 & 0xFFFFull) << 48);  // tile_dim0          (bits 127:112)
  unsigned long long q2 = 1ull                         // tile_dim1 = 1      (bits 143:128)
                        | (dim0 << 32);                // tensor_dim0_stride[31:0] (bits 207:160)
  unsigned long long q3 = 0ull;                        // stride hi / dim1_stride = 0
  tdm_i32x8 g1;
  g1[0] = (int)(unsigned)q0; g1[1] = (int)(unsigned)(q0 >> 32);
  g1[2] = (int)(unsigned)q1; g1[3] = (int)(unsigned)(q1 >> 32);
  g1[4] = (int)(unsigned)q2; g1[5] = (int)(unsigned)(q2 >> 32);
  g1[6] = (int)(unsigned)q3; g1[7] = (int)(unsigned)(q3 >> 32);

  tdm_i32x4 z4;
  z4[0] = 0; z4[1] = 0; z4[2] = 0; z4[3] = 0;
  tdm_i32x8 z8;
#pragma unroll
  for (int i = 0; i < 8; ++i) z8[i] = 0;
  __builtin_amdgcn_tensor_load_to_lds(g0, g1, z4, z4, z8, 0);
}
#endif

// ---------------------------------------------------------------------------
// Weight repack: [COUT][CIN_IN][3][3] f32  ->  [9][COUT][CIN_OUT] f16,
// scaled by folded BN scale; bias' = (b - m) * s + beta (or b if no BN).
// ---------------------------------------------------------------------------
__global__ void gmm_pack_w_kernel(const float* __restrict__ w,
                                  const float* __restrict__ bconv,
                                  const float* __restrict__ g,
                                  const float* __restrict__ be,
                                  const float* __restrict__ m,
                                  const float* __restrict__ v,
                                  _Float16* __restrict__ wpk,
                                  float* __restrict__ bout,
                                  int COUT, int CIN_IN, int CIN_OUT,
                                  int has_bn, int total) {
  int idx = blockIdx.x * blockDim.x + threadIdx.x;
  if (idx >= total) return;
  const int ci = idx % CIN_OUT;
  int t = idx / CIN_OUT;
  const int co = t % COUT;
  const int tap = t / COUT;  // kh*3+kw

  const float s = has_bn ? g[co] * rsqrtf(v[co] + BN_EPS) : 1.0f;
  float val = 0.0f;
  if (ci < CIN_IN) val = w[(size_t)co * CIN_IN * 9 + (size_t)ci * 9 + tap] * s;
  wpk[idx] = (_Float16)val;

  if (tap == 0 && ci == 0) {
    bout[co] = has_bn ? (bconv[co] - m[co]) * s + be[co] : bconv[co];
  }
}

// ---------------------------------------------------------------------------
// conv1: 7x7 stride 2 pad 3, Cin=3 -> Cout=64, BN+ReLU fused, direct f32.
// Input NCHW f32 [B,3,512,512]; output NHWC f16 [B,256,256,64].
// ---------------------------------------------------------------------------
__global__ void gmm_conv7_bn_relu_kernel(const float* __restrict__ img,
                                         const float* __restrict__ w,
                                         const float* __restrict__ bconv,
                                         const float* __restrict__ g,
                                         const float* __restrict__ be,
                                         const float* __restrict__ m,
                                         const float* __restrict__ v,
                                         _Float16* __restrict__ out,
                                         int total) {
  int idx = blockIdx.x * blockDim.x + threadIdx.x;
  if (idx >= total) return;
  const int co = idx & 63;
  int t = idx >> 6;
  const int ox = t & 255; t >>= 8;
  const int oy = t & 255;
  const int b = t >> 8;

  float acc = 0.0f;
#pragma unroll
  for (int c = 0; c < 3; ++c) {
    const float* ip = img + (size_t)(b * 3 + c) * 512 * 512;
    const float* wp = w + (size_t)(co * 3 + c) * 49;
    for (int kh = 0; kh < 7; ++kh) {
      const int iy = oy * 2 - 3 + kh;
      if (iy < 0 || iy >= 512) continue;
      const float* row = ip + (size_t)iy * 512;
      __builtin_prefetch(row + ox * 2, 0, 0);
#pragma unroll
      for (int kw = 0; kw < 7; ++kw) {
        const int ix = ox * 2 - 3 + kw;
        if (ix < 0 || ix >= 512) continue;
        acc += row[ix] * wp[kh * 7 + kw];
      }
    }
  }
  const float s = g[co] * rsqrtf(v[co] + BN_EPS);
  float y = (acc + bconv[co] - m[co]) * s + be[co];
  y = y > 0.0f ? y : 0.0f;
  out[idx] = (_Float16)y;  // idx == ((b*256+oy)*256+ox)*64+co
}

// ---------------------------------------------------------------------------
// Generic 3x3 pad-1 conv via WMMA implicit GEMM (tap-decomposed).
// in:  NHWC f16 [B][IH][IW][CIN];  wpk: [9][COUT][CIN] f16 (BN-folded)
// out: NHWC f16 [B][OH][OW][COUT], ReLU applied.
// Block = 8 waves sharing one 64-row m-group; weights for the current tap are
// staged into LDS by the TDM (double-buffered, TENSORcnt-synchronized); each
// wave computes 64(M) x 16(N) outputs. B fragments stream from global NHWC.
// Fragment addressing per ISA 05_wmma 16-bit layouts:
//   A: lane l row m=l&15; halves 0-7 -> K=kb..kb+7, 8-15 -> K=kb+16..kb+23 (kb=8*(l>>4))
//   B: lane l col n=l&15; same K order as A (two 16B loads in NHWC)
//   D: lane l col n=l&15; vgpr r    -> m=r+8*(l>>4)
// ---------------------------------------------------------------------------
template <int CIN, int COUT, int STRIDE>
__launch_bounds__(256)
__global__ void gmm_conv3x3_wmma_kernel(const _Float16* __restrict__ in,
                                        const _Float16* __restrict__ wpk,
                                        const float* __restrict__ bias,
                                        _Float16* __restrict__ out,
                                        int IH, int IW, int OH, int OW) {
  extern __shared__ _Float16 smem[];     // 2 x [64][CIN] tap buffers
  constexpr int SLAB = 64 * CIN;         // halves per tap buffer

  const int lane = threadIdx.x & 31;
  const int wv = threadIdx.x >> 5;       // wave id 0..7 -> n-tile
  const int nlo = lane & 15;
  const int khi = lane >> 4;
  const int mbase = (int)blockIdx.y << 6;

  const int ntx = OW >> 4;
  int tile = (int)blockIdx.x * 8 + wv;
  const int xb = (tile % ntx) << 4; tile /= ntx;
  const int oy = tile % OH;
  const int b = tile / OH;
  const int ox = xb + nlo;

  f8 acc[4];
#pragma unroll
  for (int mt = 0; mt < 4; ++mt)
#pragma unroll
    for (int r = 0; r < 8; ++r) acc[mt][r] = 0.0f;

  const _Float16* wbase = wpk + (size_t)mbase * CIN;  // m-group slab, tap 0

#if GMM_USE_TDM
  if (threadIdx.x < 32) {               // wave 0 drives the DMA
    tdm_load_1d(wbase, 0u, SLAB);
    __builtin_amdgcn_s_wait_tensorcnt(0);
  }
  __syncthreads();
#endif

#pragma unroll 1
  for (int tap = 0; tap < 9; ++tap) {
    const int kh = tap / 3;
    const int kw = tap % 3;
#if GMM_USE_TDM
    const int buf = tap & 1;
    if (tap < 8 && threadIdx.x < 32) {  // prefetch next tap into other buffer
      tdm_load_1d(wbase + (size_t)(tap + 1) * COUT * CIN,
                  (unsigned)((buf ^ 1) * SLAB * 2), SLAB);
    }
#else
    const int buf = 0;
    __syncthreads();
    {
      const uint4* src = (const uint4*)(wbase + (size_t)tap * COUT * CIN);
      uint4* dst = (uint4*)smem;
      for (int i = threadIdx.x; i < SLAB / 8; i += 256) dst[i] = src[i];
    }
    __syncthreads();
#endif

    const int iy = oy * STRIDE - 1 + kh;
    const bool vy = (iy >= 0) && (iy < IH);
    const int ix = ox * STRIDE - 1 + kw;
    const bool vv = vy && (ix >= 0) && (ix < IW);
    const _Float16* bp =
        in + ((size_t)(b * IH + iy) * IW + ix) * CIN + (khi << 3);
    __builtin_prefetch(bp, 0, 0);
    const _Float16* as = smem + buf * SLAB;  // [64][CIN] tap slab in LDS

#pragma unroll
    for (int ck = 0; ck < CIN; ck += 32) {
      h16 bf;
      if (vv) { const h8 blo = *(const h8*)(bp + ck); const h8 bhi = *(const h8*)(bp + ck + 16); for (int i = 0; i < 8; ++i) { bf[i] = blo[i]; bf[8 + i] = bhi[i]; } }
      else    bf = h16_zero();
#pragma unroll
      for (int mt = 0; mt < 4; ++mt) {
        const _Float16* a0 = as + ((mt << 4) + nlo) * CIN + ck + (khi << 3);
        const h8 alo = *(const h8*)a0;          // ds_load_b128
        const h8 ahi = *(const h8*)(a0 + 16);   // ds_load_b128
        h16 af;
#pragma unroll
        for (int i = 0; i < 8; ++i) { af[i] = alo[i]; af[8 + i] = ahi[i]; }
        acc[mt] = __builtin_amdgcn_wmma_f32_16x16x32_f16(
            false, af, false, bf, (short)0, acc[mt], false, false);
      }
    }

#if GMM_USE_TDM
    if (tap < 8) {
      if (threadIdx.x < 32) __builtin_amdgcn_s_wait_tensorcnt(0);
      __syncthreads();  // next-tap slab visible to all waves; old slab dead
    }
#endif
  }

  _Float16* op = out + ((size_t)(b * OH + oy) * OW + ox) * COUT;
#pragma unroll
  for (int mt = 0; mt < 4; ++mt) {
    const int co0 = mbase + (mt << 4) + (khi << 3);
    const f8 bv = *(const f8*)(bias + co0);
    h8 o;
#pragma unroll
    for (int r = 0; r < 8; ++r) {
      float x = acc[mt][r] + bv[r];
      x = x > 0.0f ? x : 0.0f;
      o[r] = (_Float16)x;
    }
    *(h8*)(op + co0) = o;  // 8 consecutive channels -> one b128 store
  }
}

// ---------------------------------------------------------------------------
// Correlation: 81 shifts (D=4); NHWC makes each map a contiguous 256-ch dot.
// Output channel-padded to 96 (81..95 = 0) so fw1 runs as Cin=96 WMMA GEMM.
// ---------------------------------------------------------------------------
__global__ void gmm_corr_kernel(const _Float16* __restrict__ cf,
                                const _Float16* __restrict__ pf,
                                _Float16* __restrict__ corr) {
  const int t = threadIdx.x;  // 0..95
  int p = blockIdx.x;
  const int x = p & 63; p >>= 6;
  const int y = p & 63;
  const int b = p >> 6;

  float acc = 0.0f;
  if (t < 81) {
    const int dy = t / 9 - 4;
    const int dx = t % 9 - 4;
    const int py = y + dy, px = x + dx;
    if (py >= 0 && py < 64 && px >= 0 && px < 64) {
      const h8* a = (const h8*)(cf + ((size_t)((b * 64 + y) * 64 + x)) * 256);
      const h8* q = (const h8*)(pf + ((size_t)((b * 64 + py) * 64 + px)) * 256);
#pragma unroll 4
      for (int c = 0; c < 32; ++c) {
        const h8 av = a[c];
        const h8 qv = q[c];
#pragma unroll
        for (int i = 0; i < 8; ++i) acc += (float)av[i] * (float)qv[i];
      }
    }
  }
  corr[(size_t)blockIdx.x * 96 + t] = (_Float16)acc;
}

// ---------------------------------------------------------------------------
// fw3: 3x3 pad 1, 64 -> 2 channels, f32 out, NCHW layout (final output).
// ---------------------------------------------------------------------------
__global__ void gmm_conv3x3_out_kernel(const _Float16* __restrict__ in,
                                       const float* __restrict__ w,
                                       const float* __restrict__ bias,
                                       float* __restrict__ out, int total) {
  int idx = blockIdx.x * blockDim.x + threadIdx.x;
  if (idx >= total) return;
  const int x = idx & 63;
  int t = idx >> 6;
  const int y = t & 63; t >>= 6;
  const int co = t & 1;
  const int b = t >> 1;

  float acc = bias[co];
#pragma unroll
  for (int kh = 0; kh < 3; ++kh) {
    const int iy = y - 1 + kh;
    if (iy < 0 || iy >= 64) continue;
#pragma unroll
    for (int kw = 0; kw < 3; ++kw) {
      const int ix = x - 1 + kw;
      if (ix < 0 || ix >= 64) continue;
      const _Float16* ip = in + ((size_t)((b * 64 + iy) * 64 + ix)) * 64;
      const float* wp = w + (size_t)co * 64 * 9 + kh * 3 + kw;
      for (int ci = 0; ci < 64; ++ci) acc += (float)ip[ci] * wp[ci * 9];
    }
  }
  out[idx] = acc;  // NCHW: ((b*2+co)*64+y)*64+x == idx
}

// ---------------------------------------------------------------------------
extern "C" void kernel_launch(void* const* d_in, const int* in_sizes, int n_in,
                              void* d_out, int out_size, void* d_ws, size_t ws_size,
                              hipStream_t stream) {
  (void)n_in; (void)out_size; (void)ws_size;
  const float* cloth  = (const float*)d_in[0];
  const float* person = (const float*)d_in[1];
  const float* w1  = (const float*)d_in[2];  const float* b1  = (const float*)d_in[3];
  const float* g1  = (const float*)d_in[4];  const float* be1 = (const float*)d_in[5];
  const float* m1  = (const float*)d_in[6];  const float* v1  = (const float*)d_in[7];
  const float* w2  = (const float*)d_in[8];  const float* b2  = (const float*)d_in[9];
  const float* g2  = (const float*)d_in[10]; const float* be2 = (const float*)d_in[11];
  const float* m2  = (const float*)d_in[12]; const float* v2  = (const float*)d_in[13];
  const float* w3  = (const float*)d_in[14]; const float* b3  = (const float*)d_in[15];
  const float* g3  = (const float*)d_in[16]; const float* be3 = (const float*)d_in[17];
  const float* m3  = (const float*)d_in[18]; const float* v3  = (const float*)d_in[19];
  const float* fw1 = (const float*)d_in[20]; const float* fb1 = (const float*)d_in[21];
  const float* fw2 = (const float*)d_in[22]; const float* fb2 = (const float*)d_in[23];
  const float* fw3 = (const float*)d_in[24]; const float* fb3 = (const float*)d_in[25];

  const int NB = in_sizes[0] / (3 * 512 * 512);  // 8

  char* ws = (char*)d_ws;
  size_t off = 0;
  auto wsa = [&](size_t bytes) -> char* {
    char* p = ws + off;
    off += (bytes + 255) & ~(size_t)255;
    return p;
  };
  _Float16* buf1 = (_Float16*)wsa((size_t)NB * 256 * 256 * 64 * 2);   // feat1 / fw1-out
  _Float16* buf2 = (_Float16*)wsa((size_t)NB * 128 * 128 * 128 * 2);  // feat2 / fw2-out
  _Float16* fc3  = (_Float16*)wsa((size_t)NB * 64 * 64 * 256 * 2);
  _Float16* fp3  = (_Float16*)wsa((size_t)NB * 64 * 64 * 256 * 2);
  _Float16* corb = (_Float16*)wsa((size_t)NB * 64 * 64 * 96 * 2);
  _Float16* wpk2  = (_Float16*)wsa(9 * 128 * 64 * 2);  float* bb2  = (float*)wsa(128 * 4);
  _Float16* wpk3  = (_Float16*)wsa(9 * 256 * 128 * 2); float* bb3  = (float*)wsa(256 * 4);
  _Float16* wpkf1 = (_Float16*)wsa(9 * 128 * 96 * 2);  float* bbf1 = (float*)wsa(128 * 4);
  _Float16* wpkf2 = (_Float16*)wsa(9 * 64 * 128 * 2);  float* bbf2 = (float*)wsa(64 * 4);
  _Float16* f1o = buf1;  // reuse after feat1 consumed
  _Float16* f2o = buf2;

  // --- weight packing (BN folded) ---
  {
    int tot = 9 * 128 * 64;
    gmm_pack_w_kernel<<<tot / 256, 256, 0, stream>>>(w2, b2, g2, be2, m2, v2,
                                                     wpk2, bb2, 128, 64, 64, 1, tot);
    tot = 9 * 256 * 128;
    gmm_pack_w_kernel<<<tot / 256, 256, 0, stream>>>(w3, b3, g3, be3, m3, v3,
                                                     wpk3, bb3, 256, 128, 128, 1, tot);
    tot = 9 * 128 * 96;
    gmm_pack_w_kernel<<<tot / 256, 256, 0, stream>>>(fw1, fb1, nullptr, nullptr, nullptr,
                                                     nullptr, wpkf1, bbf1, 128, 81, 96, 0, tot);
    tot = 9 * 64 * 128;
    gmm_pack_w_kernel<<<tot / 256, 256, 0, stream>>>(fw2, fb2, nullptr, nullptr, nullptr,
                                                     nullptr, wpkf2, bbf2, 64, 128, 128, 0, tot);
  }

  const int tot1 = NB * 256 * 256 * 64;
  // grid.x = n-tiles/8 (8 waves per block, one tile each); grid.y = m-groups
  const dim3 g_c2(NB * 128 * (128 / 16) / 8, 128 / 64);
  const dim3 g_c3(NB * 64 * (64 / 16) / 8, 256 / 64);
  const dim3 g_f1(NB * 64 * (64 / 16) / 8, 128 / 64);
  const dim3 g_f2(NB * 64 * (64 / 16) / 8, 64 / 64);
  // dynamic LDS: 2 tap buffers x 64 rows x CIN halves x 2B
  const size_t lds_c2 = 2ull * 64 * 64 * 2;
  const size_t lds_c3 = 2ull * 64 * 128 * 2;
  const size_t lds_f1 = 2ull * 64 * 96 * 2;
  const size_t lds_f2 = 2ull * 64 * 128 * 2;

  // --- feature extraction: cloth ---
  gmm_conv7_bn_relu_kernel<<<tot1 / 256, 256, 0, stream>>>(
      cloth, w1, b1, g1, be1, m1, v1, buf1, tot1);
  gmm_conv3x3_wmma_kernel<64, 128, 2><<<g_c2, 256, lds_c2, stream>>>(
      buf1, wpk2, bb2, buf2, 256, 256, 128, 128);
  gmm_conv3x3_wmma_kernel<128, 256, 2><<<g_c3, 256, lds_c3, stream>>>(
      buf2, wpk3, bb3, fc3, 128, 128, 64, 64);

  // --- feature extraction: person (reuses buf1/buf2) ---
  gmm_conv7_bn_relu_kernel<<<tot1 / 256, 256, 0, stream>>>(
      person, w1, b1, g1, be1, m1, v1, buf1, tot1);
  gmm_conv3x3_wmma_kernel<64, 128, 2><<<g_c2, 256, lds_c2, stream>>>(
      buf1, wpk2, bb2, buf2, 256, 256, 128, 128);
  gmm_conv3x3_wmma_kernel<128, 256, 2><<<g_c3, 256, lds_c3, stream>>>(
      buf2, wpk3, bb3, fp3, 128, 128, 64, 64);

  // --- correlation (81 shifts, padded to 96 ch) ---
  gmm_corr_kernel<<<NB * 64 * 64, 96, 0, stream>>>(fc3, fp3, corb);

  // --- regression head ---
  gmm_conv3x3_wmma_kernel<96, 128, 1><<<g_f1, 256, lds_f1, stream>>>(
      corb, wpkf1, bbf1, f1o, 64, 64, 64, 64);
  gmm_conv3x3_wmma_kernel<128, 64, 1><<<g_f2, 256, lds_f2, stream>>>(
      f1o, wpkf2, bbf2, f2o, 64, 64, 64, 64);
  const int tot3 = NB * 2 * 64 * 64;
  gmm_conv3x3_out_kernel<<<tot3 / 256, 256, 0, stream>>>(
      f2o, fw3, fb3, (float*)d_out, tot3);
}
